// Llama4MoE_71846212928055
// MI455X (gfx1250) — matrix operand order repaired and checked by reading, not compiled, not run on hardware
//
#include <hip/hip_runtime.h>
#include <hip/hip_bf16.h>

// Llama4-style MoE: shared SwiGLU expert + top-1 routed expert (E=8),
// grouped-GEMM formulation with bf16 WMMA on gfx1250 (wave32).
#define T_TOK 2048
#define HDIM  1024
#define FDIM  2048
#define NEXP  8
#define LDSB  40   // padded LDS row stride (bf16 elems) = 80B, breaks bank conflicts

typedef __bf16 bf16;
typedef __attribute__((ext_vector_type(16))) __bf16 bf16x16;
typedef __attribute__((ext_vector_type(8)))  __bf16 bf16x8;
typedef __attribute__((ext_vector_type(8)))  float  f32x8;
typedef __attribute__((ext_vector_type(4)))  float  f32x4;
typedef __attribute__((ext_vector_type(4)))  int    i32x4;

#if defined(__HIP_DEVICE_COMPILE__) && __has_builtin(__builtin_amdgcn_global_load_async_to_lds_b128)
#define USE_ASYNC 1
#else
#define USE_ASYNC 0
#endif

union FragBF { bf16x16 v; bf16x8 h[2]; };

// 16-byte copy global->LDS; async (ASYNCcnt-tracked) when available.
// Builtin signature (from hipcc diagnostic): param0 = int4 in device AS (global
// src), param1 = int4 in shared AS (LDS dst), then imm offset, imm cpol.
__device__ __forceinline__ void cp16(bf16* l, const bf16* g) {
#if USE_ASYNC
  typedef __attribute__((address_space(1))) i32x4 gv4_t;
  typedef __attribute__((address_space(3))) i32x4 lv4_t;
  __builtin_amdgcn_global_load_async_to_lds_b128((gv4_t*)g, (lv4_t*)l, 0, 0);
#else
  *(bf16x8*)l = *(const bf16x8*)g;
#endif
}

__device__ __forceinline__ void cp_commit_wait() {
#if USE_ASYNC
#if __has_builtin(__builtin_amdgcn_s_wait_asynccnt)
  __builtin_amdgcn_s_wait_asynccnt(0);
#else
  asm volatile("s_wait_asynccnt 0x0" ::: "memory");
#endif
#endif
  __syncthreads();
}

// A fragment 16x32 bf16 (ISA 7.12.2): lanes0-15 row=lane, v0..3=K0..7, v4..7=K16..23;
// lanes16-31 same row, K8..15 / K24..31.  LDS tile: [rows][LDSB] row-major, K-contig.
__device__ __forceinline__ bf16x16 load_frag_a(const bf16* s, int mrow, int lane) {
  int r = lane & 15, hf = lane >> 4;
  const bf16* base = s + (size_t)(mrow + r) * LDSB + hf * 8;
  FragBF f;
  f.h[0] = *(const bf16x8*)(base);
  f.h[1] = *(const bf16x8*)(base + 16);
  return f.v;
}

// B fragment 32x16 bf16: lanes0-15 col=lane K0..7 / K16..23, lanes16-31 col=lane-16 K8..15 / K24..31.
// LDS tile: [ncols][LDSB] with K contiguous per output-channel row.
__device__ __forceinline__ bf16x16 load_frag_b(const bf16* s, int ncol, int lane) {
  int c = lane & 15, kh = (lane >> 4) * 8;
  const bf16* base = s + (size_t)(ncol + c) * LDSB + kh;
  FragBF f;
  f.h[0] = *(const bf16x8*)(base);
  f.h[1] = *(const bf16x8*)(base + 16);
  return f.v;
}

__device__ __forceinline__ f32x8 wmma_bf16(bf16x16 a, bf16x16 b, f32x8 c) {
  return __builtin_amdgcn_wmma_f32_16x16x32_bf16(false, a, false, b, (short)0, c,
                                                 false, false);
}

__device__ __forceinline__ float silu(float x) {
  return x / (1.f + __expf(-x));
}

// ---------------- phase 0: weight fp32 -> bf16, zero counters ----------------
__global__ void k_prep(const float* __restrict__ s0, const float* __restrict__ s1,
                       const float* __restrict__ s2, const float* __restrict__ s3,
                       const float* __restrict__ s4, const float* __restrict__ s5,
                       bf16* __restrict__ dst, int* __restrict__ cnt) {
  if (blockIdx.x == 0 && threadIdx.x < NEXP) cnt[threadIdx.x] = 0;
  const size_t FH = (size_t)FDIM * HDIM;
  const size_t b1 = FH, b2 = 2 * FH, b3 = 3 * FH, b4 = 11 * FH, b5 = 19 * FH,
               tot = 27 * FH;
  size_t stride = (size_t)gridDim.x * blockDim.x * 4;
  for (size_t i = ((size_t)blockIdx.x * blockDim.x + threadIdx.x) * 4; i < tot;
       i += stride) {
    const float* s; size_t o;
    if      (i < b1) { s = s0; o = i;      }
    else if (i < b2) { s = s1; o = i - b1; }
    else if (i < b3) { s = s2; o = i - b2; }
    else if (i < b4) { s = s3; o = i - b3; }
    else if (i < b5) { s = s4; o = i - b4; }
    else             { s = s5; o = i - b5; }
    f32x4 v = *(const f32x4*)(s + o);
    dst[i + 0] = (bf16)v.x; dst[i + 1] = (bf16)v.y;
    dst[i + 2] = (bf16)v.z; dst[i + 3] = (bf16)v.w;
  }
}

// ---------------- phase 1: router (one wave per token) ----------------
__global__ void k_router(const float* __restrict__ x, const float* __restrict__ gw,
                         bf16* __restrict__ xbf, bf16* __restrict__ xsbf,
                         int* __restrict__ top, int* __restrict__ cnt) {
  int lane = threadIdx.x & 31;
  int t = blockIdx.x * 8 + (threadIdx.x >> 5);
  const float* xr = x + (size_t)t * HDIM;
  float acc[NEXP];
#pragma unroll
  for (int e = 0; e < NEXP; e++) acc[e] = 0.f;
  for (int h = lane; h < HDIM; h += 32) {
    float xv = xr[h];
#pragma unroll
    for (int e = 0; e < NEXP; e++) acc[e] += xv * gw[e * HDIM + h];
  }
#pragma unroll
  for (int e = 0; e < NEXP; e++)
    for (int o = 16; o > 0; o >>= 1) acc[e] += __shfl_xor(acc[e], o, 32);
  int best = 0; float m = acc[0];
#pragma unroll
  for (int e = 1; e < NEXP; e++) if (acc[e] > m) { m = acc[e]; best = e; }
  float score = 1.f / (1.f + __expf(-m));
  if (lane == 0) { top[t] = best; atomicAdd(&cnt[best], 1); }
  bf16* xb = xbf + (size_t)t * HDIM;
  bf16* xs = xsbf + (size_t)t * HDIM;
  for (int h = lane; h < HDIM; h += 32) {
    float v = xr[h];
    xb[h] = (bf16)v;
    xs[h] = (bf16)(v * score);
  }
}

// ---------------- phase 2: scan + scatter (expert-sorted permutation) --------
__global__ void k_scan(const int* __restrict__ cnt, int* __restrict__ offs,
                       int* __restrict__ cur) {
  if (threadIdx.x == 0) {
    int a = 0;
    for (int e = 0; e < NEXP; e++) { offs[e] = a; cur[e] = a; a += cnt[e]; }
    offs[NEXP] = a;
  }
}

__global__ void k_scatter(const int* __restrict__ top, int* __restrict__ cur,
                          int* __restrict__ perm) {
  int t = blockIdx.x * blockDim.x + threadIdx.x;
  if (t < T_TOK) {
    int e = top[t];
    int p = atomicAdd(&cur[e], 1);
    perm[p] = t;
  }
}

// ---------------- phase 3: gate+up GEMM pair + SiLU (bf16 WMMA) --------------
// C tile 64(M) x 128(N), 8 waves of 32x32, K-steps of 32 over HDIM.
__global__ __launch_bounds__(256)
void k_gateup(const bf16* __restrict__ A, const bf16* __restrict__ Wg,
              const bf16* __restrict__ Wu, const int* __restrict__ perm,
              const int* __restrict__ offs, bf16* __restrict__ Y, int wstride) {
  __shared__ __align__(16) bf16 sA[64 * LDSB];
  __shared__ __align__(16) bf16 sG[128 * LDSB];
  __shared__ __align__(16) bf16 sU[128 * LDSB];
  int tid = threadIdx.x, lane = tid & 31, wid = tid >> 5;
  int z = blockIdx.z;
  int rbase = 0, rend = T_TOK;
  if (offs) { rbase = offs[z]; rend = offs[z + 1]; }
  int m0 = rbase + blockIdx.y * 64;
  if (m0 >= rend) return;                 // block-uniform: EXEC stays full
  const bf16* wg = Wg + (size_t)z * wstride;
  const bf16* wu = Wu + (size_t)z * wstride;
  int n0 = blockIdx.x * 128;

  // copy addressing: A = 1x b128/thread, each B = 2x b128/thread
  int ar = tid >> 2, aseg = tid & 3;
  int prow = m0 + ar; if (prow > rend - 1) prow = rend - 1;   // clamp: row-local garbage
  int tok = perm ? perm[prow] : prow;
  const bf16* gA = A + (size_t)tok * HDIM + aseg * 8;
  bf16* lA = sA + ar * LDSB + aseg * 8;
  const bf16* gG = wg + (size_t)(n0 + ar) * HDIM + aseg * 8;
  const bf16* gU = wu + (size_t)(n0 + ar) * HDIM + aseg * 8;
  bf16* lG = sG + ar * LDSB + aseg * 8;
  bf16* lU = sU + ar * LDSB + aseg * 8;

  int wm = wid >> 2, wn = wid & 3;
  const f32x8 zero = {0.f, 0.f, 0.f, 0.f, 0.f, 0.f, 0.f, 0.f};
  f32x8 accG[2][2], accU[2][2];
#pragma unroll
  for (int mi = 0; mi < 2; mi++)
#pragma unroll
    for (int ni = 0; ni < 2; ni++) { accG[mi][ni] = zero; accU[mi][ni] = zero; }

  for (int k0 = 0; k0 < HDIM; k0 += 32) {
    __syncthreads();
    cp16(lA, gA + k0);
    cp16(lG, gG + k0);
    cp16(lG + 64 * LDSB, gG + (size_t)64 * HDIM + k0);
    cp16(lU, gU + k0);
    cp16(lU + 64 * LDSB, gU + (size_t)64 * HDIM + k0);
    cp_commit_wait();
    bf16x16 a0 = load_frag_a(sA, wm * 32, lane);
    bf16x16 a1 = load_frag_a(sA, wm * 32 + 16, lane);
#pragma unroll
    for (int ni = 0; ni < 2; ni++) {
      bf16x16 bg = load_frag_b(sG, wn * 32 + ni * 16, lane);
      bf16x16 bu = load_frag_b(sU, wn * 32 + ni * 16, lane);
      accG[0][ni] = wmma_bf16(a0, bg, accG[0][ni]);
      accG[1][ni] = wmma_bf16(a1, bg, accG[1][ni]);
      accU[0][ni] = wmma_bf16(a0, bu, accU[0][ni]);
      accU[1][ni] = wmma_bf16(a1, bu, accU[1][ni]);
    }
  }

  // epilogue: y = silu(g) * u, bf16.  C layout: M=(lane>>4)*8+j, N=lane&15.
  int rb = (lane >> 4) * 8, cb = lane & 15;
#pragma unroll
  for (int mi = 0; mi < 2; mi++)
#pragma unroll
    for (int ni = 0; ni < 2; ni++) {
      int col = n0 + wn * 32 + ni * 16 + cb;
#pragma unroll
      for (int j = 0; j < 8; j++) {
        int p = m0 + wm * 32 + mi * 16 + rb + j;
        if (p < rend) {
          float y = silu(accG[mi][ni][j]) * accU[mi][ni][j];
          Y[(size_t)p * FDIM + col] = (bf16)y;
        }
      }
    }
}

// ---------------- phase 4: down GEMM, write or scatter-add fp32 --------------
__global__ __launch_bounds__(256)
void k_down(const bf16* __restrict__ Yb, const bf16* __restrict__ Wd,
            const int* __restrict__ perm, const int* __restrict__ offs,
            float* __restrict__ out, int wstride, int accum) {
  __shared__ __align__(16) bf16 sA[64 * LDSB];
  __shared__ __align__(16) bf16 sB[128 * LDSB];
  int tid = threadIdx.x, lane = tid & 31, wid = tid >> 5;
  int z = blockIdx.z;
  int rbase = 0, rend = T_TOK;
  if (offs) { rbase = offs[z]; rend = offs[z + 1]; }
  int m0 = rbase + blockIdx.y * 64;
  if (m0 >= rend) return;
  const bf16* wd = Wd + (size_t)z * wstride;
  int n0 = blockIdx.x * 128;

  int ar = tid >> 2, aseg = tid & 3;
  int prow = m0 + ar; if (prow > rend - 1) prow = rend - 1;
  const bf16* gA = Yb + (size_t)prow * FDIM + aseg * 8;   // Y rows are segment rows
  bf16* lA = sA + ar * LDSB + aseg * 8;
  const bf16* gB = wd + (size_t)(n0 + ar) * FDIM + aseg * 8;
  bf16* lB = sB + ar * LDSB + aseg * 8;

  int wm = wid >> 2, wn = wid & 3;
  const f32x8 zero = {0.f, 0.f, 0.f, 0.f, 0.f, 0.f, 0.f, 0.f};
  f32x8 acc[2][2];
#pragma unroll
  for (int mi = 0; mi < 2; mi++)
#pragma unroll
    for (int ni = 0; ni < 2; ni++) acc[mi][ni] = zero;

  for (int k0 = 0; k0 < FDIM; k0 += 32) {
    __syncthreads();
    cp16(lA, gA + k0);
    cp16(lB, gB + k0);
    cp16(lB + 64 * LDSB, gB + (size_t)64 * FDIM + k0);
    cp_commit_wait();
    bf16x16 a0 = load_frag_a(sA, wm * 32, lane);
    bf16x16 a1 = load_frag_a(sA, wm * 32 + 16, lane);
#pragma unroll
    for (int ni = 0; ni < 2; ni++) {
      bf16x16 b = load_frag_b(sB, wn * 32 + ni * 16, lane);
      acc[0][ni] = wmma_bf16(a0, b, acc[0][ni]);
      acc[1][ni] = wmma_bf16(a1, b, acc[1][ni]);
    }
  }

  int rb = (lane >> 4) * 8, cb = lane & 15;
#pragma unroll
  for (int mi = 0; mi < 2; mi++)
#pragma unroll
    for (int ni = 0; ni < 2; ni++) {
      int col = n0 + wn * 32 + ni * 16 + cb;
#pragma unroll
      for (int j = 0; j < 8; j++) {
        int p = m0 + wm * 32 + mi * 16 + rb + j;
        if (p < rend) {
          int tk = perm ? perm[p] : p;
          size_t o = (size_t)tk * HDIM + col;
          float v = acc[mi][ni][j];
          out[o] = accum ? (out[o] + v) : v;
        }
      }
    }
}

// ---------------- launcher ----------------
extern "C" void kernel_launch(void* const* d_in, const int* in_sizes, int n_in,
                              void* d_out, int out_size, void* d_ws, size_t ws_size,
                              hipStream_t stream) {
  const float* x  = (const float*)d_in[0];   // [T,H]
  const float* gw = (const float*)d_in[1];   // [E,H]
  const float* sg = (const float*)d_in[2];   // [F,H]
  const float* su = (const float*)d_in[3];   // [F,H]
  const float* sd = (const float*)d_in[4];   // [H,F]
  const float* rg = (const float*)d_in[5];   // [E,F,H]
  const float* ru = (const float*)d_in[6];   // [E,F,H]
  const float* rd = (const float*)d_in[7];   // [E,H,F]
  float* out = (float*)d_out;

  const size_t FH = (size_t)FDIM * HDIM;
  bf16* wsb = (bf16*)d_ws;
  bf16 *WgS = wsb,           *WuS = wsb + FH,      *WdS = wsb + 2 * FH;
  bf16 *WgR = wsb + 3 * FH,  *WuR = wsb + 11 * FH, *WdR = wsb + 19 * FH;
  bf16 *Xbf = wsb + 27 * FH, *Xsb = wsb + 28 * FH, *Y   = wsb + 29 * FH; // Y: [T,F]
  int* meta = (int*)(wsb + 31 * FH);
  int *top = meta, *perm = top + T_TOK, *cnt = perm + T_TOK;
  int *offs = cnt + NEXP, *cur = offs + NEXP + 1;

  k_prep<<<2048, 256, 0, stream>>>(sg, su, sd, rg, ru, rd, wsb, cnt);
  k_router<<<T_TOK / 8, 256, 0, stream>>>(x, gw, Xbf, Xsb, top, cnt);
  k_scan<<<1, 32, 0, stream>>>(cnt, offs, cur);
  k_scatter<<<T_TOK / 256, 256, 0, stream>>>(top, cur, perm);

  // shared expert: all tokens
  k_gateup<<<dim3(FDIM / 128, T_TOK / 64, 1), 256, 0, stream>>>(
      Xbf, WgS, WuS, nullptr, nullptr, Y, 0);
  k_down<<<dim3(HDIM / 128, T_TOK / 64, 1), 256, 0, stream>>>(
      Y, WdS, nullptr, nullptr, out, 0, 0);

  // routed experts: grouped GEMM over expert-sorted segments (early-exit tiles)
  k_gateup<<<dim3(FDIM / 128, T_TOK / 64, NEXP), 256, 0, stream>>>(
      Xsb, WgR, WuR, perm, offs, Y, (int)FH);
  k_down<<<dim3(HDIM / 128, T_TOK / 64, NEXP), 256, 0, stream>>>(
      Y, WdR, perm, offs, out, (int)FH, 1);
}
